// GroundedWorldModel_18751827214786
// MI455X (gfx1250) — matrix operand order repaired and checked by reading, not compiled, not run on hardware
//
#include <hip/hip_runtime.h>

typedef __attribute__((ext_vector_type(16))) _Float16 v16h;
typedef __attribute__((ext_vector_type(8)))  _Float16 v8h;
typedef __attribute__((ext_vector_type(8)))  float    v8f;

#define SHUF16(lo,hi) __builtin_shufflevector((lo),(hi),0,1,2,3,4,5,6,7,8,9,10,11,12,13,14,15)
#define WMMA_F16(A,B,C) __builtin_amdgcn_wmma_f32_16x16x32_f16(false,(A),false,(B),(short)0,(C),false,false)

// Problem constants
#define NPIX   16384      // 128*128
#define OBSD   768
#define OD     128        // OBJ_DIM
#define NS     32         // N_SLOTS
#define HIDD   256
#define RELD   64
#define RENDIN 69632      // OD*NS + RELD*NS*NS
#define RH     1536       // 2*OBS_DIM
#define NBLK_ATTN 128     // pixel blocks of 128
#define NBLK_R1   512     // renderer layer-1 row blocks (69632/512 = 136 rows each)

// ---------------- workspace layout (bytes) ----------------
#define OFF_KPACK   ((size_t)0)               // 8*24*512 halves = 192KB
#define OFF_VPACK   ((size_t)196608)
#define OFF_KF16    ((size_t)393216)          // 16384*128 halves = 4MB
#define OFF_VF16    ((size_t)4587520)
#define OFF_QH      ((size_t)8781824)         // 32*128 halves
#define OFF_SLOTS   ((size_t)8790016)         // 32*128 f32
#define OFF_U       ((size_t)8806400)         // 32*128 f32
#define OFF_ASUM    ((size_t)8823040)         // 32 f32 (padded)
#define OFF_OBJ     ((size_t)8823296)         // 32*128 f32
#define OFF_FEAT    ((size_t)8839680)         // 69632 f32
#define OFF_UPART   ((size_t)9118208)         // 128*4096 f32 = 2MB
#define OFF_APART   ((size_t)11215360)        // 128*32 f32
#define OFF_HPART   ((size_t)11231744)        // 512*1536 f32 = 3MB
#define OFF_HACC    ((size_t)14377472)        // 1536 f32

// ===== K0: pack Wk/Wv (fp32 768x128) into f16 WMMA-B fragment order =====
// frag f = ntile*24 + kk ; within frag: lane l (0..31) x half h (0..15)
// B[K][N] with K = 32*kk + h + 16*(l>>4), N = 16*ntile + (l&15)
__global__ __launch_bounds__(256) void pack_w_kernel(const float* __restrict__ Wk,
                                                     const float* __restrict__ Wv,
                                                     _Float16* __restrict__ kp,
                                                     _Float16* __restrict__ vp) {
  int idx = blockIdx.x * 256 + threadIdx.x;     // 0 .. 2*98304
  int mtx = idx / 98304;
  int t   = idx % 98304;
  int f = t >> 9, w = t & 511, l = w >> 4, h = w & 15;
  int ntile = f / 24, kk = f % 24;
  int kd = 32 * kk + h + 16 * (l >> 4);
  int nc = 16 * ntile + (l & 15);
  float val = (mtx == 0 ? Wk : Wv)[kd * OD + nc];
  (mtx == 0 ? kp : vp)[t] = (_Float16)val;
}

// ===== K1: LayerNorm(obs) then k = x@Wk, v = x@Wv via WMMA (f16 in, f32 acc)
__global__ __launch_bounds__(256) void lnkv_kernel(const float* __restrict__ obs,
                                                   const float* __restrict__ g,
                                                   const float* __restrict__ b,
                                                   const _Float16* __restrict__ kp,
                                                   const _Float16* __restrict__ vp,
                                                   _Float16* __restrict__ kf,
                                                   _Float16* __restrict__ vf) {
  __shared__ float    xf[16][OBSD];
  __shared__ __align__(16) _Float16 xh[16][OBSD];
  __shared__ float r1[16][16], r2[16][16], mst[16][2];
  int tid = threadIdx.x, row0 = blockIdx.x * 16;
  int r = tid >> 4, sub = tid & 15;

  const float* src = obs + (size_t)(row0 + r) * OBSD;
  float s1 = 0.f, s2 = 0.f;
  for (int j = sub; j < OBSD; j += 16) { float v = src[j]; xf[r][j] = v; s1 += v; s2 += v * v; }
  r1[r][sub] = s1; r2[r][sub] = s2;
  __syncthreads();
  if (sub == 0) {
    float a = 0.f, c = 0.f;
    for (int i = 0; i < 16; i++) { a += r1[r][i]; c += r2[r][i]; }
    float mean = a / (float)OBSD, var = c / (float)OBSD - mean * mean;
    mst[r][0] = mean; mst[r][1] = rsqrtf(var + 1e-5f);
  }
  __syncthreads();
  float mean = mst[r][0], rstd = mst[r][1];
  for (int j = sub; j < OBSD; j += 16)
    xh[r][j] = (_Float16)((xf[r][j] - mean) * rstd * g[j] + b[j]);
  __syncthreads();

  int wv = tid >> 5, lane = tid & 31, m = lane & 15, hi = lane >> 4;
  v8f accK = {0.f,0.f,0.f,0.f,0.f,0.f,0.f,0.f};
  v8f accV = {0.f,0.f,0.f,0.f,0.f,0.f,0.f,0.f};
  for (int kk = 0; kk < 24; kk++) {
    int kb = kk * 32;
    v16h a = SHUF16(*(const v8h*)&xh[m][kb + 16 * hi],
                    *(const v8h*)&xh[m][kb + 16 * hi + 8]);
    const _Float16* bpk = kp + ((size_t)(wv * 24 + kk) << 9) + lane * 16;
    const _Float16* bpv = vp + ((size_t)(wv * 24 + kk) << 9) + lane * 16;
    v16h bk = SHUF16(*(const v8h*)bpk, *(const v8h*)(bpk + 8));
    v16h bv = SHUF16(*(const v8h*)bpv, *(const v8h*)(bpv + 8));
    accK = WMMA_F16(a, bk, accK);
    accV = WMMA_F16(a, bv, accV);
  }
  #pragma unroll
  for (int rr = 0; rr < 8; rr++) {
    int row = row0 + rr + 8 * hi, col = wv * 16 + m;
    kf[(size_t)row * OD + col] = (_Float16)accK[rr];
    vf[(size_t)row * OD + col] = (_Float16)accV[rr];
  }
}

// ===== K2: slots = mu + exp(log_sigma) * noise
__global__ __launch_bounds__(256) void init_slots_kernel(const float* __restrict__ mu,
                                                         const float* __restrict__ ls,
                                                         const float* __restrict__ noise,
                                                         float* __restrict__ slots) {
  for (int o = threadIdx.x; o < NS * OD; o += 256) {
    int d = o & 127;
    slots[o] = mu[d] + expf(ls[d]) * noise[o];
  }
}

// ===== K3: q = LN(slots) @ Wq  (fp32 compute, f16 output for WMMA A)
__global__ __launch_bounds__(256) void qproj_kernel(const float* __restrict__ slots,
                                                    const float* __restrict__ gg,
                                                    const float* __restrict__ bb,
                                                    const float* __restrict__ Wq,
                                                    _Float16* __restrict__ qh) {
  __shared__ float sln[NS][OD];
  __shared__ float r1[NS][8], r2[NS][8], mst[NS][2];
  int tid = threadIdx.x;
  int r = tid >> 3, sub = tid & 7;
  float s1 = 0.f, s2 = 0.f;
  for (int j = sub; j < OD; j += 8) { float v = slots[r * OD + j]; sln[r][j] = v; s1 += v; s2 += v * v; }
  r1[r][sub] = s1; r2[r][sub] = s2;
  __syncthreads();
  if (sub == 0) {
    float a = 0.f, c = 0.f;
    for (int i = 0; i < 8; i++) { a += r1[r][i]; c += r2[r][i]; }
    float mean = a / (float)OD, var = c / (float)OD - mean * mean;
    mst[r][0] = mean; mst[r][1] = rsqrtf(var + 1e-5f);
  }
  __syncthreads();
  { float mean = mst[r][0], rstd = mst[r][1];
    for (int j = sub; j < OD; j += 8) sln[r][j] = (sln[r][j] - mean) * rstd * gg[j] + bb[j]; }
  __syncthreads();
  for (int o = tid; o < NS * OD; o += 256) {
    int s = o >> 7, d = o & 127;
    float acc = 0.f;
    for (int c = 0; c < OD; c++) acc += sln[s][c] * Wq[c * OD + d];
    qh[o] = (_Float16)acc;
  }
}

// ===== K4: attention block kernel (WMMA logits + softmax + WMMA attn@v) =====
// 128 blocks x 128 pixels. Writes deterministic per-block partials.
__global__ __launch_bounds__(256) void attn_kernel(const _Float16* __restrict__ qh,
                                                   const _Float16* __restrict__ kf,
                                                   const _Float16* __restrict__ vf,
                                                   float* __restrict__ Upart,
                                                   float* __restrict__ apart) {
  __shared__ __align__(16) _Float16 ql[NS * OD];       // 8KB  q (slot-major)
  __shared__ __align__(16) _Float16 kt[128 * OD];      // 32KB k tile [pixel][feat]
  __shared__ __align__(16) _Float16 vT[OD * 128];      // 32KB v tile transposed [feat][pixel]
  __shared__ float    lg[128 * NS];                    // 16KB logits -> attn [pixel][slot]
  __shared__ __align__(16) _Float16 at[NS * 128];      // 8KB  attn f16 [slot][pixel]
  int tid = threadIdx.x, p0 = blockIdx.x * 128;

  ((uint4*)ql)[tid] = ((const uint4*)qh)[tid];         // 2048 halves = 256 x 16B (slots 0..15 only)
  for (int i = tid; i < 2048; i += 256)
    ((uint4*)kt)[i] = ((const uint4*)(kf + (size_t)p0 * OD))[i];
  for (int i = tid; i < 2048; i += 256) {
    union { uint4 u; _Float16 h[8]; } tmp;
    tmp.u = ((const uint4*)(vf + (size_t)p0 * OD))[i];
    int pix = i >> 4, f0 = (i & 15) * 8;
    #pragma unroll
    for (int j = 0; j < 8; j++) vT[(f0 + j) * 128 + pix] = tmp.h[j];
  }
  __syncthreads();

  int wv = tid >> 5, lane = tid & 31, m = lane & 15, hi = lane >> 4;
  int np0 = wv * 16;
  const float scale = 0.08838834764831845f;  // 128^-0.5

  // logits[slot][pixel] = q @ k^T
  for (int mt = 0; mt < 2; mt++) {
    v8f acc = {0.f,0.f,0.f,0.f,0.f,0.f,0.f,0.f};
    int M = m + 16 * mt;
    for (int kk = 0; kk < 4; kk++) {
      int kb = 32 * kk;
      v16h a = SHUF16(*(const v8h*)&ql[M * OD + kb + 16 * hi],
                      *(const v8h*)&ql[M * OD + kb + 16 * hi + 8]);
      const _Float16* bp = &kt[(np0 + m) * OD + kb + 16 * hi];
      v16h bfr = SHUF16(*(const v8h*)bp, *(const v8h*)(bp + 8));
      acc = WMMA_F16(a, bfr, acc);
    }
    #pragma unroll
    for (int rr = 0; rr < 8; rr++)
      lg[(np0 + m) * NS + (16 * mt + rr + 8 * hi)] = acc[rr] * scale;
  }
  __syncthreads();

  // softmax over slots (axis=1 of reference) per pixel; overwrite lg with attn
  if (tid < 128) {
    float vals[NS];
    float mx = -1e30f;
    for (int s = 0; s < NS; s++) { float v = lg[tid * NS + s]; vals[s] = v; mx = fmaxf(mx, v); }
    float sum = 0.f;
    for (int s = 0; s < NS; s++) { float e = expf(vals[s] - mx); vals[s] = e; sum += e; }
    float inv = 1.f / sum;
    for (int s = 0; s < NS; s++) {
      float a = vals[s] * inv;
      lg[tid * NS + s] = a;
      at[s * 128 + tid] = (_Float16)a;
    }
  }
  __syncthreads();

  // deterministic per-slot attention mass for this block
  if (tid < NS) {
    float t = 0.f;
    for (int p = 0; p < 128; p++) t += lg[p * NS + tid];
    apart[blockIdx.x * NS + tid] = t;
  }

  // U_partial = attn @ v   (A = attn [32 x 128pix], B = v [128pix x 128dim])
  int dt = wv;  // this wave's 16-dim output tile
  for (int mt = 0; mt < 2; mt++) {
    v8f acc = {0.f,0.f,0.f,0.f,0.f,0.f,0.f,0.f};
    int M = m + 16 * mt;
    for (int kk = 0; kk < 4; kk++) {
      int kb = 32 * kk;
      v16h a = SHUF16(*(const v8h*)&at[M * 128 + kb + 16 * hi],
                      *(const v8h*)&at[M * 128 + kb + 16 * hi + 8]);
      const _Float16* bp = &vT[(16 * dt + m) * 128 + kb + 16 * hi];
      v16h bfr = SHUF16(*(const v8h*)bp, *(const v8h*)(bp + 8));
      acc = WMMA_F16(a, bfr, acc);
    }
    #pragma unroll
    for (int rr = 0; rr < 8; rr++) {
      int slot = 16 * mt + rr + 8 * hi, dim = 16 * dt + m;
      Upart[(size_t)blockIdx.x * (NS * OD) + slot * OD + dim] = acc[rr];
    }
  }
}

// ===== K5: reduce attention partials (deterministic)
__global__ __launch_bounds__(256) void ureduce_kernel(const float* __restrict__ Upart,
                                                      const float* __restrict__ apart,
                                                      float* __restrict__ U,
                                                      float* __restrict__ asum) {
  int e = blockIdx.x * 256 + threadIdx.x;  // 4096
  float s = 0.f;
  for (int b = 0; b < NBLK_ATTN; b++) s += Upart[(size_t)b * (NS * OD) + e];
  U[e] = s;
  if (e < NS) {
    float t = 0.f;
    for (int b = 0; b < NBLK_ATTN; b++) t += apart[b * NS + e];
    asum[e] = t;
  }
}

// ===== K6: slots += U/(sum+eps); then slots += MLP(LN(slots))
__global__ __launch_bounds__(256) void slotup_kernel(float* __restrict__ slots,
                                                     const float* __restrict__ U,
                                                     const float* __restrict__ asum,
                                                     const float* __restrict__ gm,
                                                     const float* __restrict__ bm,
                                                     const float* __restrict__ w1,
                                                     const float* __restrict__ b1,
                                                     const float* __restrict__ w2,
                                                     const float* __restrict__ b2) {
  __shared__ float sl[NS][OD];
  __shared__ float hln[NS][OD];
  __shared__ float hid[NS][HIDD];
  __shared__ float r1[NS][8], r2[NS][8], mst[NS][2];
  int tid = threadIdx.x;
  for (int o = tid; o < NS * OD; o += 256) {
    int s = o >> 7;
    float v = slots[o] + U[o] / (asum[s] + 1e-8f);
    slots[o] = v; sl[s][o & 127] = v;
  }
  __syncthreads();
  int r = tid >> 3, sub = tid & 7;
  float s1 = 0.f, s2 = 0.f;
  for (int j = sub; j < OD; j += 8) { float v = sl[r][j]; s1 += v; s2 += v * v; }
  r1[r][sub] = s1; r2[r][sub] = s2;
  __syncthreads();
  if (sub == 0) {
    float a = 0.f, c = 0.f;
    for (int i = 0; i < 8; i++) { a += r1[r][i]; c += r2[r][i]; }
    float mean = a / (float)OD, var = c / (float)OD - mean * mean;
    mst[r][0] = mean; mst[r][1] = rsqrtf(var + 1e-5f);
  }
  __syncthreads();
  { float mean = mst[r][0], rstd = mst[r][1];
    for (int j = sub; j < OD; j += 8) hln[r][j] = (sl[r][j] - mean) * rstd * gm[j] + bm[j]; }
  __syncthreads();
  for (int o = tid; o < NS * HIDD; o += 256) {
    int s = o >> 8, h = o & 255;
    float acc = b1[h];
    for (int c = 0; c < OD; c++) acc += hln[s][c] * w1[c * HIDD + h];
    hid[s][h] = fmaxf(acc, 0.f);
  }
  __syncthreads();
  for (int o = tid; o < NS * OD; o += 256) {
    int s = o >> 7, d = o & 127;
    float acc = b2[d];
    for (int c = 0; c < HIDD; c++) acc += hid[s][c] * w2[c * OD + d];
    slots[o] += acc;
  }
}

// ===== K7: objects = prop MLP(slots); dynamics -> next_objects into feat[0:4096]
__global__ __launch_bounds__(256) void objdyn_kernel(const float* __restrict__ slots,
                                                     const float* __restrict__ pw1, const float* __restrict__ pb1,
                                                     const float* __restrict__ pw2, const float* __restrict__ pb2,
                                                     const float* __restrict__ action,
                                                     const float* __restrict__ dw1, const float* __restrict__ db1,
                                                     const float* __restrict__ dw2, const float* __restrict__ db2,
                                                     const float* __restrict__ dw3, const float* __restrict__ db3,
                                                     const float* __restrict__ gravity,
                                                     float* __restrict__ objects,
                                                     float* __restrict__ feat) {
  __shared__ float sl[NS][OD];
  __shared__ float ph[NS][HIDD];
  __shared__ float obj[NS][OD];
  __shared__ float h2[NS][HIDD];
  int tid = threadIdx.x;
  for (int o = tid; o < NS * OD; o += 256) sl[o >> 7][o & 127] = slots[o];
  __syncthreads();
  for (int o = tid; o < NS * HIDD; o += 256) {
    int s = o >> 8, h = o & 255;
    float acc = pb1[h];
    for (int c = 0; c < OD; c++) acc += sl[s][c] * pw1[c * HIDD + h];
    ph[s][h] = fmaxf(acc, 0.f);
  }
  __syncthreads();
  for (int o = tid; o < NS * OD; o += 256) {
    int s = o >> 7, d = o & 127;
    float acc = pb2[d];
    for (int c = 0; c < HIDD; c++) acc += ph[s][c] * pw2[c * OD + d];
    obj[s][d] = acc; objects[o] = acc;
  }
  __syncthreads();
  for (int o = tid; o < NS * HIDD; o += 256) {
    int s = o >> 8, h = o & 255;
    float acc = db1[h];
    for (int c = 0; c < OD; c++) acc += obj[s][c] * dw1[c * HIDD + h];
    for (int c = 0; c < 16; c++) acc += action[c] * dw1[(OD + c) * HIDD + h];
    ph[s][h] = fmaxf(acc, 0.f);
  }
  __syncthreads();
  for (int o = tid; o < NS * HIDD; o += 256) {
    int s = o >> 8, h = o & 255;
    float acc = db2[h];
    for (int c = 0; c < HIDD; c++) acc += ph[s][c] * dw2[c * HIDD + h];
    h2[s][h] = fmaxf(acc, 0.f);
  }
  __syncthreads();
  for (int o = tid; o < NS * OD; o += 256) {
    int s = o >> 7, d = o & 127;
    float acc = db3[d];
    for (int c = 0; c < HIDD; c++) acc += h2[s][c] * dw3[c * OD + d];
    if (d < 3) acc += gravity[d] * 0.1f;
    feat[o] = obj[s][d] + acc * 0.1f;          // next_objects flattened
  }
}

// ===== K8: pairwise relations -> feat[4096:]
__global__ __launch_bounds__(128) void rel_kernel(const float* __restrict__ objects,
                                                  const float* __restrict__ w1, const float* __restrict__ b1,
                                                  const float* __restrict__ w2, const float* __restrict__ b2,
                                                  float* __restrict__ feat) {
  __shared__ float pin[2 * OD];
  __shared__ float hid[128];
  int pair = blockIdx.x, i = pair >> 5, j = pair & 31, tid = threadIdx.x;
  pin[tid]       = objects[i * OD + tid];
  pin[OD + tid]  = objects[j * OD + tid];
  __syncthreads();
  float acc = b1[tid];
  for (int c = 0; c < 2 * OD; c++) acc += pin[c] * w1[c * 128 + tid];
  hid[tid] = fmaxf(acc, 0.f);
  __syncthreads();
  if (tid < RELD) {
    float a2 = b2[tid];
    for (int c = 0; c < 128; c++) a2 += hid[c] * w2[c * RELD + tid];
    feat[NS * OD + pair * RELD + tid] = a2;
  }
}

// ===== K9: renderer layer 1, streaming 428MB weight matvec (HBM-bound) =====
// Speculative prefetch (global_prefetch_b8) runs ~16 rows (96KB/block-slice)
// ahead of the FMA stream to keep the 23.3 TB/s HBM pipe full.
__global__ __launch_bounds__(256) void rend1_kernel(const float* __restrict__ feat,
                                                    const float* __restrict__ w1,
                                                    float* __restrict__ hpart) {
  __shared__ float fch[136];
  int tid = threadIdx.x, r0 = blockIdx.x * 136;   // 512*136 == 69632
  if (tid < 136) fch[tid] = feat[r0 + tid];
  __syncthreads();
  float acc[6] = {0.f, 0.f, 0.f, 0.f, 0.f, 0.f};
  for (int rr = 0; rr < 136; rr++) {
    if ((rr & 7) == 0) {
      int pr = r0 + rr + 16;
      if (pr < RENDIN) __builtin_prefetch(w1 + (size_t)pr * RH + tid, 0, 1);
    }
    float f = fch[rr];
    const float* wr = w1 + (size_t)(r0 + rr) * RH + tid;
    #pragma unroll
    for (int j = 0; j < 6; j++) acc[j] += f * wr[256 * j];
  }
  #pragma unroll
  for (int j = 0; j < 6; j++) hpart[(size_t)blockIdx.x * RH + tid + 256 * j] = acc[j];
}

// ===== K10: reduce renderer partials
__global__ __launch_bounds__(256) void hreduce_kernel(const float* __restrict__ hpart,
                                                      float* __restrict__ hacc) {
  int h = blockIdx.x * 256 + threadIdx.x;  // 1536
  float s = 0.f;
  for (int b = 0; b < NBLK_R1; b++) s += hpart[(size_t)b * RH + h];
  hacc[h] = s;
}

// ===== K11: out = relu(h + b1) @ rend_w2 + b2
__global__ __launch_bounds__(256) void rend2_kernel(const float* __restrict__ hacc,
                                                    const float* __restrict__ b1,
                                                    const float* __restrict__ w2,
                                                    const float* __restrict__ b2,
                                                    float* __restrict__ out) {
  int o = blockIdx.x * 256 + threadIdx.x;  // 768
  float acc = b2[o];
  for (int h = 0; h < RH; h++) {
    if ((h & 31) == 0 && h + 32 < RH) __builtin_prefetch(w2 + (size_t)(h + 32) * OBSD + o, 0, 1);
    float a = fmaxf(hacc[h] + b1[h], 0.f);
    acc += a * w2[h * OBSD + o];
  }
  out[o] = acc;
}

extern "C" void kernel_launch(void* const* d_in, const int* in_sizes, int n_in,
                              void* d_out, int out_size, void* d_ws, size_t ws_size,
                              hipStream_t stream) {
  (void)in_sizes; (void)n_in; (void)out_size; (void)ws_size;
  const float* obs       = (const float*)d_in[0];
  const float* action    = (const float*)d_in[1];
  const float* noise     = (const float*)d_in[2];
  const float* slots_mu  = (const float*)d_in[3];
  const float* slots_ls  = (const float*)d_in[4];
  const float* ln_in_g   = (const float*)d_in[5];
  const float* ln_in_b   = (const float*)d_in[6];
  const float* ln_s_g    = (const float*)d_in[7];
  const float* ln_s_b    = (const float*)d_in[8];
  const float* ln_m_g    = (const float*)d_in[9];
  const float* ln_m_b    = (const float*)d_in[10];
  const float* Wq        = (const float*)d_in[11];
  const float* Wk        = (const float*)d_in[12];
  const float* Wv        = (const float*)d_in[13];
  const float* mlp_w1    = (const float*)d_in[14];
  const float* mlp_b1    = (const float*)d_in[15];
  const float* mlp_w2    = (const float*)d_in[16];
  const float* mlp_b2    = (const float*)d_in[17];
  const float* prop_w1   = (const float*)d_in[18];
  const float* prop_b1   = (const float*)d_in[19];
  const float* prop_w2   = (const float*)d_in[20];
  const float* prop_b2   = (const float*)d_in[21];
  const float* rel_w1    = (const float*)d_in[22];
  const float* rel_b1    = (const float*)d_in[23];
  const float* rel_w2    = (const float*)d_in[24];
  const float* rel_b2    = (const float*)d_in[25];
  const float* dyn_w1    = (const float*)d_in[26];
  const float* dyn_b1    = (const float*)d_in[27];
  const float* dyn_w2    = (const float*)d_in[28];
  const float* dyn_b2    = (const float*)d_in[29];
  const float* dyn_w3    = (const float*)d_in[30];
  const float* dyn_b3    = (const float*)d_in[31];
  const float* gravity   = (const float*)d_in[32];
  const float* rend_w1   = (const float*)d_in[33];
  const float* rend_b1   = (const float*)d_in[34];
  const float* rend_w2   = (const float*)d_in[35];
  const float* rend_b2   = (const float*)d_in[36];

  char* ws = (char*)d_ws;
  _Float16* kpack = (_Float16*)(ws + OFF_KPACK);
  _Float16* vpack = (_Float16*)(ws + OFF_VPACK);
  _Float16* kf16  = (_Float16*)(ws + OFF_KF16);
  _Float16* vf16  = (_Float16*)(ws + OFF_VF16);
  _Float16* qh    = (_Float16*)(ws + OFF_QH);
  float* slots    = (float*)(ws + OFF_SLOTS);
  float* U        = (float*)(ws + OFF_U);
  float* asum     = (float*)(ws + OFF_ASUM);
  float* objects  = (float*)(ws + OFF_OBJ);
  float* feat     = (float*)(ws + OFF_FEAT);
  float* Upart    = (float*)(ws + OFF_UPART);
  float* apart    = (float*)(ws + OFF_APART);
  float* hpart    = (float*)(ws + OFF_HPART);
  float* hacc     = (float*)(ws + OFF_HACC);

  pack_w_kernel<<<768, 256, 0, stream>>>(Wk, Wv, kpack, vpack);
  lnkv_kernel<<<NPIX / 16, 256, 0, stream>>>(obs, ln_in_g, ln_in_b, kpack, vpack, kf16, vf16);
  init_slots_kernel<<<1, 256, 0, stream>>>(slots_mu, slots_ls, noise, slots);

  for (int it = 0; it < 3; ++it) {
    qproj_kernel<<<1, 256, 0, stream>>>(slots, ln_s_g, ln_s_b, Wq, qh);
    attn_kernel<<<NBLK_ATTN, 256, 0, stream>>>(qh, kf16, vf16, Upart, apart);
    ureduce_kernel<<<16, 256, 0, stream>>>(Upart, apart, U, asum);
    slotup_kernel<<<1, 256, 0, stream>>>(slots, U, asum, ln_m_g, ln_m_b,
                                         mlp_w1, mlp_b1, mlp_w2, mlp_b2);
  }

  objdyn_kernel<<<1, 256, 0, stream>>>(slots, prop_w1, prop_b1, prop_w2, prop_b2,
                                       action, dyn_w1, dyn_b1, dyn_w2, dyn_b2,
                                       dyn_w3, dyn_b3, gravity, objects, feat);
  rel_kernel<<<NS * NS, 128, 0, stream>>>(objects, rel_w1, rel_b1, rel_w2, rel_b2, feat);
  rend1_kernel<<<NBLK_R1, 256, 0, stream>>>(feat, rend_w1, hpart);
  hreduce_kernel<<<RH / 256, 256, 0, stream>>>(hpart, hacc);
  rend2_kernel<<<OBSD / 256, 256, 0, stream>>>(hacc, rend_b1, rend_w2, rend_b2, (float*)d_out);
}
